// Feature_Extractor_43705587204338
// MI455X (gfx1250) — matrix operand order repaired and checked by reading, not compiled, not run on hardware
//
#include <hip/hip_runtime.h>
#include <hip/hip_bf16.h>

typedef __attribute__((ext_vector_type(16))) __bf16 v16bf;
typedef __attribute__((ext_vector_type(8)))  __bf16 bf16x8;
typedef __attribute__((ext_vector_type(8)))  float  v8f;

#define L_SEQ    200
#define DIM      128
#define HDIM     512
#define LTILES   13         // ceil(200/16)
#define A_STRIDE 264        // 256 + 8 bf16 pad -> 528B row stride (16B aligned, bank-skewed)
#define BPB      4          // batches per block: amortizes the W-fragment register preload

// Gather one 16-float chunk of a concatenated embedding row and convert to bf16.
__device__ __forceinline__ void gather_cvt(const int* __restrict__ x,
                                           const int* __restrict__ ids,
                                           const float* __restrict__ rat_emd,
                                           const float* __restrict__ id_emd,
                                           int bOff, int l, int tChk,
                                           bf16x8& h0, bf16x8& h1)
{
  h0 = {};
  h1 = {};
  if (l < L_SEQ) {
    const float* src = (tChk < 8)
        ? rat_emd + (long)x[bOff + l]   * DIM + tChk * 16
        : id_emd  + (long)ids[bOff + l] * DIM + (tChk - 8) * 16;
    const float4* s4 = (const float4*)src;
    float4 f0 = s4[0], f1 = s4[1], f2 = s4[2], f3 = s4[3];
    h0[0]=(__bf16)f0.x; h0[1]=(__bf16)f0.y; h0[2]=(__bf16)f0.z; h0[3]=(__bf16)f0.w;
    h0[4]=(__bf16)f1.x; h0[5]=(__bf16)f1.y; h0[6]=(__bf16)f1.z; h0[7]=(__bf16)f1.w;
    h1[0]=(__bf16)f2.x; h1[1]=(__bf16)f2.y; h1[2]=(__bf16)f2.z; h1[3]=(__bf16)f2.w;
    h1[4]=(__bf16)f3.x; h1[5]=(__bf16)f3.y; h1[6]=(__bf16)f3.z; h1[7]=(__bf16)f3.w;
  }
}

__global__ __launch_bounds__(256)
void feat_extract_wmma(const int* __restrict__ x,
                       const int* __restrict__ ids,
                       const float* __restrict__ rat_emd,
                       const float* __restrict__ id_emd,
                       const float* __restrict__ W,
                       const float* __restrict__ bias,
                       float* __restrict__ out,
                       int B)
{
  __shared__ alignas(16) __bf16 sA[2][16][A_STRIDE];   // double-buffered gathered A tile

  const int nChunk = blockIdx.x;          // 0..3 (128 cols each)
  const int lane   = threadIdx.x & 31;
  const int wave   = threadIdx.x >> 5;    // 0..7, each wave owns a 16-col N tile
  const int nCol   = (nChunk << 7) + (wave << 4) + (lane & 15);

  // ---- Preload B fragments (W columns) as bf16 into registers; reused for BPB*13 tiles.
  // Dense 16-bit B (32x16): lanes 0-15 hold K=0-7 & 16-23, lanes 16-31 hold K=8-15 & 24-31,
  // two K values packed per VGPR -> v16bf element j == K = kbase + j + (j >= 8 ? 8 : 0).
  v16bf Bfrag[8];
  #pragma unroll
  for (int ks = 0; ks < 8; ++ks) {
    const int kbase = ks * 32 + ((lane & 16) ? 8 : 0);
    #pragma unroll
    for (int j = 0; j < 16; ++j)
      Bfrag[ks][j] = (__bf16)W[(kbase + j + ((j & 8) ? 8 : 0)) * HDIM + nCol];
  }
  const float bcol = bias[nCol];

  const int tRow = threadIdx.x >> 4;      // staging: 0..15 (row of A tile)
  const int tChk = threadIdx.x & 15;      // staging: chunk of 16 floats

  for (int bs = 0; bs < BPB; ++bs) {
    const int b = blockIdx.y * BPB + bs;
    if (b >= B) break;
    const int bOff = b * L_SEQ;

    float colSum = 0.0f;

    // Prologue: gather tile 0 into registers.
    bf16x8 c0, c1;
    gather_cvt(x, ids, rat_emd, id_emd, bOff, tRow, tChk, c0, c1);

    for (int lt = 0; lt < LTILES; ++lt) {
      const int p = lt & 1;

      // Publish current tile to LDS buffer p.
      *(bf16x8*)&sA[p][tRow][tChk * 16]     = c0;
      *(bf16x8*)&sA[p][tRow][tChk * 16 + 8] = c1;
      __syncthreads();

      // Issue next tile's global gathers now so they overlap the WMMA chain.
      if (lt + 1 < LTILES) {
        gather_cvt(x, ids, rat_emd, id_emd, bOff, (lt + 1) * 16 + tRow, tChk, c0, c1);
      } else {
        c0 = {}; c1 = {};
        if (bs + 1 < BPB && b + 1 < B) {   // warm next batch's first gather rows
          const float* nsrc = (tChk < 8)
              ? rat_emd + (long)x[bOff + L_SEQ + tRow]   * DIM + tChk * 16
              : id_emd  + (long)ids[bOff + L_SEQ + tRow] * DIM + (tChk - 8) * 16;
          __builtin_prefetch(nsrc, 0, 1);
        }
      }

      // Preload all 8 A fragments (one ds_load clause), then 8 back-to-back WMMAs.
      // 16-bit A (16x32): M = lane&15; lanes 0-15: K 0-7 & 16-23; lanes 16-31: K 8-15 & 24-31.
      const char* aRow = (const char*)&sA[p][lane & 15][0] + ((lane & 16) ? 16 : 0);
      v16bf afr[8];
      #pragma unroll
      for (int ks = 0; ks < 8; ++ks) {
        __builtin_memcpy(&afr[ks],             aRow + ks * 64,      16);
        __builtin_memcpy((char*)&afr[ks] + 16, aRow + ks * 64 + 32, 16);
      }
      v8f acc = {};
      #pragma unroll
      for (int ks = 0; ks < 8; ++ks)
        acc = __builtin_amdgcn_wmma_f32_16x16x32_bf16(false, afr[ks], false, Bfrag[ks],
                                                      (short)0, acc, false, false);

      // Bias + ReLU + masked row reduction (rows l >= 200 must not contribute).
      // C layout: VGPR i holds M = i (lanes 0-15) / M = i+8 (lanes 16-31), N = lane&15.
      const int mBase = lt * 16 + ((lane & 16) ? 8 : 0);
      float rsum = 0.0f;
      #pragma unroll
      for (int i = 0; i < 8; ++i) {
        float v = acc[i] + bcol;
        v = v > 0.0f ? v : 0.0f;
        rsum += ((mBase + i) < L_SEQ) ? v : 0.0f;
      }
      rsum += __shfl_xor(rsum, 16, 32);   // fold the two 8-row halves (same column)
      colSum += rsum;
      // No barrier needed here: next iteration stores to the *other* buffer, and
      // this iteration's LDS reads are provably complete before any wave reaches
      // the store that reuses this buffer (WMMA consumption forces s_wait_dscnt).
    }

    if ((lane & 16) == 0)
      out[b * HDIM + nCol] = colSum * (1.0f / (float)L_SEQ);

    __syncthreads();   // close tail race before next batch overwrites buffer 0
  }
}

extern "C" void kernel_launch(void* const* d_in, const int* in_sizes, int n_in,
                              void* d_out, int out_size, void* d_ws, size_t ws_size,
                              hipStream_t stream) {
  const int*   x   = (const int*)d_in[0];
  const int*   ids = (const int*)d_in[1];
  const float* re  = (const float*)d_in[2];
  const float* ie  = (const float*)d_in[3];
  const float* W   = (const float*)d_in[4];
  const float* bv  = (const float*)d_in[5];
  float* out = (float*)d_out;

  const int B = in_sizes[0] / L_SEQ;            // 2048
  dim3 grid(HDIM / 128, (B + BPB - 1) / BPB);   // (4 N-chunks, 512 batch groups)
  feat_extract_wmma<<<grid, dim3(256), 0, stream>>>(x, ids, re, ie, W, bv, out, B);
}
